// BasicBlock_69002944577761
// MI455X (gfx1250) — 8-bit matrix operand order repaired and checked by reading, compiled, not run on hardware
//
#include <hip/hip_runtime.h>

typedef __attribute__((ext_vector_type(8))) int v8i;

#define BATCH 32
#define CH    256
#define HH    56
#define WW    56
#define HWSZ  (HH*WW)
#define MTOT  (BATCH*HWSZ)
#define NELEM ((size_t)BATCH*(size_t)CH*(size_t)HWSZ)
#define KCH   4

static __device__ __forceinline__ signed char sgn8(float v) {
    return (v > 0.f) ? (signed char)1 : ((v < 0.f) ? (signed char)-1 : (signed char)0);
}

__global__ void zero_stats_k(float* __restrict__ p) {
    p[blockIdx.x * 256 + threadIdx.x] = 0.f;
}

__global__ void bin_act_k(const float* __restrict__ x, signed char* __restrict__ a) {
    size_t i = (size_t)blockIdx.x * 256 + threadIdx.x;
    int c = (int)(i & 255);
    size_t pos = i >> 8;
    int w = (int)(pos % WW);
    size_t t2 = pos / WW;
    int h = (int)(t2 % HH);
    int b = (int)(t2 / HH);
    size_t nchw = (((size_t)b * CH + c) * HH + h) * WW + w;
    a[i] = sgn8(x[nchw]);
}

__global__ void pack_w_k(const float* __restrict__ wsrc, signed char* __restrict__ bw) {
    size_t i = (size_t)blockIdx.x * 256 + threadIdx.x;
    int k = (int)(i & 63);
    size_t t2 = i >> 6;
    int n = (int)(t2 & 255);
    size_t t3 = t2 >> 8;
    int cc = (int)(t3 & 3);
    int tap = (int)(t3 >> 2);
    int kh = tap / 3, kw = tap % 3;
    int cin = cc * 64 + k;
    float v = wsrc[(((size_t)n * CH + cin) * 3 + kh) * 3 + kw];
    bw[i] = sgn8(v);
}

__global__ __launch_bounds__(256)
void binconv_wmma_k(const signed char* __restrict__ A,
                    const signed char* __restrict__ Bw,
                    const float* __restrict__ zpad,
                    int* __restrict__ Y) {
    __shared__ uint4 SMEM[2304];

    const int t    = threadIdx.x;
    const int lane = t & 31;
    const int wid  = t >> 5;
    const int waveM = wid & 3;
    const int waveN = wid >> 2;
    const int hi   = lane >> 4;
    const int lm   = lane & 15;
    const int mBase = blockIdx.x * 128;
    const int nBase = blockIdx.y * 64;

    int ph[2], pw[2];
    const signed char* gbase[2];
    unsigned int ldsAoff[2];
    #pragma unroll
    for (int i2 = 0; i2 < 2; ++i2) {
        int s = t + i2 * 256;
        int p = s >> 2, q = s & 3;
        int m = mBase + p;
        int b = m / HWSZ;
        int rem = m - b * HWSZ;
        int h = rem / WW;
        int w = rem - h * WW;
        ph[i2] = h; pw[i2] = w;
        gbase[i2] = A + ((((size_t)b * HH + h) * WW + w) << 8) + q * 16;
        ldsAoff[i2] = (unsigned int)((p * 4 + q) * 16);
    }
    const signed char* bbase = Bw + ((size_t)nBase << 6) + (size_t)t * 16;
    const signed char* zptr  = (const signed char*)zpad;

    auto issue = [&](int it, int buf) {
        const int tap = it >> 2, cc = it & 3;
        const int dh = tap / 3 - 1, dw = (tap % 3) - 1;
        const long doff = ((long)dh * WW + dw) * 256 + cc * 64;
        char* abufp = (char*)&SMEM[buf * 512];
        char* bbufp = (char*)&SMEM[1536 + buf * 256];
        #pragma unroll
        for (int i2 = 0; i2 < 2; ++i2) {
            int hh = ph[i2] + dh, ww = pw[i2] + dw;
            const signed char* src =
                ((unsigned)hh < (unsigned)HH && (unsigned)ww < (unsigned)WW)
                    ? (gbase[i2] + doff) : zptr;
            unsigned long long ga = (unsigned long long)(uintptr_t)src;
            unsigned int lo = (unsigned int)(uintptr_t)(abufp + ldsAoff[i2]);
            asm volatile("global_load_async_to_lds_b128 %0, %1, off"
                         :: "v"(lo), "v"(ga) : "memory");
        }
        {
            unsigned long long ga =
                (unsigned long long)(uintptr_t)(bbase + (size_t)it * (CH * 64));
            unsigned int lo = (unsigned int)(uintptr_t)(bbufp + (size_t)t * 16);
            asm volatile("global_load_async_to_lds_b128 %0, %1, off"
                         :: "v"(lo), "v"(ga) : "memory");
        }
    };

    v8i acc[2][2] = {};

    issue(0, 0);
    issue(1, 1);
    #pragma unroll 3
    for (int it = 0; it < 36; ++it) {
        const int buf = it % 3;
        if (it < 35) asm volatile("s_wait_asynccnt 0x3" ::: "memory");
        else         asm volatile("s_wait_asynccnt 0x0" ::: "memory");
        __syncthreads();
        if (it + 2 < 36) issue(it + 2, (it + 2) % 3);

        const int2* A2 = (const int2*)&SMEM[buf * 512];
        const int4* B4 = (const int4*)&SMEM[1536 + buf * 256];
        v8i afrag[2], bfrag[2];
        #pragma unroll
        for (int mf = 0; mf < 2; ++mf) {
            int row = waveM * 32 + mf * 16 + lm;
            int idx = row * 8 + hi * 2;
            int2 p0 = A2[idx + 0];
            int2 p1 = A2[idx + 1];
            int2 p2 = A2[idx + 4];
            int2 p3 = A2[idx + 5];
            v8i a;
            a[0] = p0.x; a[1] = p0.y; a[2] = p1.x; a[3] = p1.y;
            a[4] = p2.x; a[5] = p2.y; a[6] = p3.x; a[7] = p3.y;
            afrag[mf] = a;
        }
        #pragma unroll
        for (int nf = 0; nf < 2; ++nf) {
            int col = waveN * 32 + nf * 16 + lm;
            int bidx = col * 4 + hi;
            int4 q0 = B4[bidx + 0];
            int4 q1 = B4[bidx + 2];
            v8i b;
            b[0] = q0.x; b[1] = q0.y; b[2] = q0.z; b[3] = q0.w;
            b[4] = q1.x; b[5] = q1.y; b[6] = q1.z; b[7] = q1.w;
            bfrag[nf] = b;
        }
        #pragma unroll
        for (int mf = 0; mf < 2; ++mf)
            #pragma unroll
            for (int nf = 0; nf < 2; ++nf)
                acc[mf][nf] = __builtin_amdgcn_wmma_i32_16x16x64_iu8(
                    true, afrag[mf], true, bfrag[nf], acc[mf][nf], false, false);
    }

    #pragma unroll
    for (int mf = 0; mf < 2; ++mf)
        #pragma unroll
        for (int nf = 0; nf < 2; ++nf) {
            int n = nBase + waveN * 32 + nf * 16 + lm;
            #pragma unroll
            for (int r = 0; r < 8; ++r) {
                int m = mBase + waveM * 32 + mf * 16 + r + hi * 8;
                Y[(size_t)m * CH + n] = acc[mf][nf][r];
            }
        }
}

__global__ void channel_stats_k(const int* __restrict__ Y,
                                float* __restrict__ sum, float* __restrict__ sumsq) {
    int ch = threadIdx.x;
    size_t mstart = (size_t)blockIdx.x * 256;
    float s = 0.f, s2 = 0.f;
    for (int r = 0; r < 256; ++r) {
        if (r + 8 < 256)
            __builtin_prefetch(&Y[(mstart + r + 8) * CH + ch], 0, 1);
        float v = (float)Y[(mstart + r) * CH + ch];
        s += v; s2 += v * v;
    }
    atomicAdd(&sum[ch], s);
    atomicAdd(&sumsq[ch], s2);
}

__global__ void bn_params_k(const float* __restrict__ sum, const float* __restrict__ sumsq,
                            const float* __restrict__ gamma, const float* __restrict__ beta,
                            float* __restrict__ scale, float* __restrict__ shift) {
    int c = threadIdx.x;
    const float inv = 1.0f / (float)MTOT;
    float mu = sum[c] * inv;
    float var = sumsq[c] * inv - mu * mu;
    float sc = gamma[c] * rsqrtf(var + 1e-5f);
    scale[c] = sc;
    shift[c] = beta[c] - mu * sc;
}

__global__ void fuse1_k(const int* __restrict__ Y,
                        const float* __restrict__ scale, const float* __restrict__ shift,
                        const float* __restrict__ x, const float* __restrict__ off,
                        float* __restrict__ out1, signed char* __restrict__ abin) {
    size_t i = (size_t)blockIdx.x * 256 + threadIdx.x;
    int c = (int)(i & 255);
    size_t pos = i >> 8;
    int w = (int)(pos % WW);
    size_t t2 = pos / WW;
    int h = (int)(t2 % HH);
    int b = (int)(t2 / HH);
    size_t nchw = (((size_t)b * CH + c) * HH + h) * WW + w;
    float v = (float)Y[i] * scale[c] + shift[c] + x[nchw] + off[nchw & 511];
    v = fminf(1.f, fmaxf(-1.f, v));
    out1[nchw] = v;
    abin[i] = sgn8(v);
}

__global__ void fuse2_k(const int* __restrict__ Y,
                        const float* __restrict__ scale, const float* __restrict__ shift,
                        const float* __restrict__ out1, const float* __restrict__ off,
                        float* __restrict__ out) {
    size_t i = (size_t)blockIdx.x * 256 + threadIdx.x;
    int c = (int)(i & 255);
    size_t pos = i >> 8;
    int w = (int)(pos % WW);
    size_t t2 = pos / WW;
    int h = (int)(t2 % HH);
    int b = (int)(t2 / HH);
    size_t nchw = (((size_t)b * CH + c) * HH + h) * WW + w;
    float v = (float)Y[i] * scale[c] + shift[c] + out1[nchw] + off[nchw & 511];
    out[nchw] = fminf(1.f, fmaxf(-1.f, v));
}

extern "C" void kernel_launch(void* const* d_in, const int* in_sizes, int n_in,
                              void* d_out, int out_size, void* d_ws, size_t ws_size,
                              hipStream_t stream) {
    const float* x     = (const float*)d_in[0];
    const float* w1    = (const float*)d_in[1];
    const float* g1    = (const float*)d_in[2];
    const float* b1    = (const float*)d_in[3];
    const float* off1  = (const float*)d_in[4];
    const float* w2    = (const float*)d_in[5];
    const float* g2    = (const float*)d_in[6];
    const float* b2    = (const float*)d_in[7];
    const float* off2  = (const float*)d_in[8];
    float* out = (float*)d_out;

    char* ws = (char*)d_ws;
    signed char* a_bin = (signed char*)(ws);
    int*   y     = (int*)  (ws + NELEM);
    float* out1  = (float*)(ws + NELEM + NELEM * 4);
    signed char* bw = (signed char*)(ws + NELEM + NELEM * 8);
    float* stats = (float*)(ws + NELEM + NELEM * 8 + 9 * 4 * 256 * 64);
    float* sum1 = stats;        float* sq1 = stats + 256;
    float* sum2 = stats + 512;  float* sq2 = stats + 768;
    float* sc1  = stats + 1024; float* sh1 = stats + 1280;
    float* sc2  = stats + 1536; float* sh2 = stats + 1792;
    float* zpad = stats + 2048;

    const int ew_blocks = (int)(NELEM / 256);
    dim3 convGrid(MTOT / 128, CH / 64);

    zero_stats_k<<<9, 256, 0, stream>>>(stats);

    bin_act_k<<<ew_blocks, 256, 0, stream>>>(x, a_bin);
    pack_w_k<<<(9 * KCH * CH * 64) / 256, 256, 0, stream>>>(w1, bw);
    binconv_wmma_k<<<convGrid, 256, 0, stream>>>(a_bin, bw, zpad, y);
    channel_stats_k<<<MTOT / 256, 256, 0, stream>>>(y, sum1, sq1);
    bn_params_k<<<1, 256, 0, stream>>>(sum1, sq1, g1, b1, sc1, sh1);
    fuse1_k<<<ew_blocks, 256, 0, stream>>>(y, sc1, sh1, x, off1, out1, a_bin);

    pack_w_k<<<(9 * KCH * CH * 64) / 256, 256, 0, stream>>>(w2, bw);
    binconv_wmma_k<<<convGrid, 256, 0, stream>>>(a_bin, bw, zpad, y);
    channel_stats_k<<<MTOT / 256, 256, 0, stream>>>(y, sum2, sq2);
    bn_params_k<<<1, 256, 0, stream>>>(sum2, sq2, g2, b2, sc2, sh2);
    fuse2_k<<<ew_blocks, 256, 0, stream>>>(y, sc2, sh2, out1, off2, out);
}
